// CausalSelfAttention_77713138253893
// MI455X (gfx1250) — matrix operand order repaired and checked by reading, not compiled, not run on hardware
//
#include <hip/hip_runtime.h>
#include <hip/hip_bf16.h>

typedef __attribute__((ext_vector_type(16))) __bf16 v16bf;
typedef __attribute__((ext_vector_type(8)))  float  v8f;
typedef __attribute__((ext_vector_type(4)))  unsigned int v4u;
typedef __attribute__((ext_vector_type(8)))  int  v8i;
typedef __attribute__((ext_vector_type(4)))  int  v4i;

#define B_  2
#define T_  2048
#define C_  1024
#define NH_ 16
#define HS_ 64
#define N3_ 3072

#if __has_builtin(__builtin_amdgcn_tensor_load_to_lds) && __has_builtin(__builtin_amdgcn_s_wait_tensorcnt)
#define USE_TDM 1
#else
#define USE_TDM 0
#endif

union FragB16 { unsigned int u[8]; v16bf v; };

__device__ inline unsigned short f32_to_bf16(float f) {
    unsigned int u = __builtin_bit_cast(unsigned int, f);
    u += 0x7FFFu + ((u >> 16) & 1u);      // round-to-nearest-even
    return (unsigned short)(u >> 16);
}

__device__ inline v8f wmma_bf16(v16bf a, v16bf b, v8f c) {
    return __builtin_amdgcn_wmma_f32_16x16x32_bf16(false, a, false, b, (short)0, c, false, false);
}

#if USE_TDM
// Issue one TDM 2-D tile load: (128 rows x 32 elems) of bf16, row stride = stride0 elems.
__device__ inline void tdm_load_tile(unsigned lds_off, const void* gaddr,
                                     int tensor_d0, int tensor_d1, int stride0) {
    unsigned long long ga = (unsigned long long)(size_t)gaddr;
    v4u g0;
    g0[0] = 1u;                                              // count=1, user mode
    g0[1] = lds_off;                                         // LDS byte address
    g0[2] = (unsigned)(ga & 0xFFFFFFFFu);                    // global_addr[31:0]
    g0[3] = (unsigned)((ga >> 32) & 0x01FFFFFFu) | 0x80000000u; // addr[56:32] | type=2
    v8i g1;
    g1[0] = 1 << 16;                                         // data_size = 2 bytes
    g1[1] = (tensor_d0 & 0xFFFF) << 16;                      // tensor_dim0[15:0]
    g1[2] = ((tensor_d0 >> 16) & 0xFFFF) | ((tensor_d1 & 0xFFFF) << 16);
    g1[3] = ((tensor_d1 >> 16) & 0xFFFF) | (32 << 16);       // tile_dim0 = 32
    g1[4] = 128;                                             // tile_dim1 = 128, tile_dim2 = 0
    g1[5] = stride0;                                         // tensor_dim0_stride[31:0]
    g1[6] = 0;
    g1[7] = 0;
    v4i zz4 = {0, 0, 0, 0};
    v8i zz8 = {0, 0, 0, 0, 0, 0, 0, 0};
    __builtin_amdgcn_tensor_load_to_lds(g0, g1, zz4, zz4, zz8, 0);
}
#endif

// ---------------- conversion kernels ----------------
__global__ void conv_f32_bf16(const float* __restrict__ in,
                              unsigned short* __restrict__ out, int n) {
    int i = blockIdx.x * blockDim.x + threadIdx.x;
    if (i < n) out[i] = f32_to_bf16(in[i]);
}

// in: (K, N) row-major fp32  ->  out: (N, K) row-major bf16 (i.e. column-major)
__global__ void conv_transpose_bf16(const float* __restrict__ in,
                                    unsigned short* __restrict__ out, int K, int N) {
    int i = blockIdx.x * blockDim.x + threadIdx.x;
    if (i < K * N) {
        int k = i / N, n = i - k * N;
        out[n * K + k] = f32_to_bf16(in[i]);
    }
}

// ---------------- tiled GEMM mainloop ----------------
// Block tile 128(M) x 128(N), K-chunks of 32, double-buffered LDS.
// 8 waves in 2(M) x 4(N); wave tile 64x32 -> acc[4][2] 16x16 tiles.
// A: (M,K) bf16 row-major; Bm: (N,K) bf16 row-major (i.e. B column-major).
__device__ __forceinline__ void gemm_mainloop(
    const unsigned short* __restrict__ A, const unsigned short* __restrict__ Bm,
    int row0, int col0, int Kdim, int Mdim, int Ndim, v8f acc[4][2]) {
    __shared__ unsigned short As[2][128 * 32];
    __shared__ unsigned short Bs[2][128 * 32];
    const int tid  = threadIdx.x;
    const int lane = tid & 31;
    const int wave = tid >> 5;
    const int hl   = lane >> 4;
    const int l16  = lane & 15;
    const int wm   = wave >> 2;      // 0..1
    const int wn   = wave & 3;       // 0..3
    const int nch  = Kdim / 32;
    (void)Mdim; (void)Ndim;

#if USE_TDM
    if (tid < 32) {
        tdm_load_tile((unsigned)(size_t)&As[0][0], A + (size_t)row0 * Kdim, Kdim, Mdim, Kdim);
        tdm_load_tile((unsigned)(size_t)&Bs[0][0], Bm + (size_t)col0 * Kdim, Kdim, Ndim, Kdim);
    }
    for (int c = 0; c < nch; ++c) {
        __syncthreads();                       // buffer (c+1)&1 free of readers
        if (tid < 32) {
            if (c + 1 < nch) {
                int k1 = (c + 1) * 32;
                tdm_load_tile((unsigned)(size_t)&As[(c + 1) & 1][0],
                              A + (size_t)row0 * Kdim + k1, Kdim, Mdim, Kdim);
                tdm_load_tile((unsigned)(size_t)&Bs[(c + 1) & 1][0],
                              Bm + (size_t)col0 * Kdim + k1, Kdim, Ndim, Kdim);
                __builtin_amdgcn_s_wait_tensorcnt(2);   // chunk c landed
            } else {
                __builtin_amdgcn_s_wait_tensorcnt(0);
            }
        }
        __syncthreads();                       // chunk c visible to all waves
#else
    const int r  = tid >> 1;
    const int hh = tid & 1;
    uint4 ra0, ra1, rb0, rb1;
    {
        const unsigned short* gpa = A + (size_t)(row0 + r) * Kdim + hh * 16;
        ra0 = *(const uint4*)gpa; ra1 = *(const uint4*)(gpa + 8);
        const unsigned short* gpb = Bm + (size_t)(col0 + r) * Kdim + hh * 16;
        rb0 = *(const uint4*)gpb; rb1 = *(const uint4*)(gpb + 8);
    }
    for (int c = 0; c < nch; ++c) {
        __syncthreads();                       // buffer c&1 free of readers
        {
            unsigned short* la = &As[c & 1][r * 32 + hh * 16];
            *(uint4*)la = ra0; *(uint4*)(la + 8) = ra1;
            unsigned short* lb = &Bs[c & 1][r * 32 + hh * 16];
            *(uint4*)lb = rb0; *(uint4*)(lb + 8) = rb1;
        }
        if (c + 1 < nch) {
            int k1 = (c + 1) * 32;
            const unsigned short* gpa = A + (size_t)(row0 + r) * Kdim + k1 + hh * 16;
            ra0 = *(const uint4*)gpa; ra1 = *(const uint4*)(gpa + 8);
            const unsigned short* gpb = Bm + (size_t)(col0 + r) * Kdim + k1 + hh * 16;
            rb0 = *(const uint4*)gpb; rb1 = *(const uint4*)(gpb + 8);
        }
        __syncthreads();                       // chunk c visible
#endif
        const unsigned short* Ab = As[c & 1];
        const unsigned short* Bb = Bs[c & 1];
        FragB16 af[4], bfr[2];
#pragma unroll
        for (int mi = 0; mi < 4; ++mi) {
            const unsigned short* pa = Ab + (wm * 64 + 16 * mi + l16) * 32 + 8 * hl;
            uint4 lo = *(const uint4*)pa;
            uint4 hi = *(const uint4*)(pa + 16);
            af[mi].u[0] = lo.x; af[mi].u[1] = lo.y; af[mi].u[2] = lo.z; af[mi].u[3] = lo.w;
            af[mi].u[4] = hi.x; af[mi].u[5] = hi.y; af[mi].u[6] = hi.z; af[mi].u[7] = hi.w;
        }
#pragma unroll
        for (int nj = 0; nj < 2; ++nj) {
            const unsigned short* pb = Bb + (wn * 32 + 16 * nj + l16) * 32 + 8 * hl;
            uint4 lo = *(const uint4*)pb;
            uint4 hi = *(const uint4*)(pb + 16);
            bfr[nj].u[0] = lo.x; bfr[nj].u[1] = lo.y; bfr[nj].u[2] = lo.z; bfr[nj].u[3] = lo.w;
            bfr[nj].u[4] = hi.x; bfr[nj].u[5] = hi.y; bfr[nj].u[6] = hi.z; bfr[nj].u[7] = hi.w;
        }
#pragma unroll
        for (int mi = 0; mi < 4; ++mi)
#pragma unroll
            for (int nj = 0; nj < 2; ++nj)
                acc[mi][nj] = wmma_bf16(af[mi].v, bfr[nj].v, acc[mi][nj]);
    }
}

// ---------------- QKV GEMM: (4096 x 1024) @ (1024 x 3072) ----------------
// grid = (3072/128, 4096/128), block = 256
__global__ __launch_bounds__(256) void qkv_gemm(
    const unsigned short* __restrict__ xb,
    const unsigned short* __restrict__ WaT,
    const float* __restrict__ b_attn,
    unsigned short* __restrict__ Qb,           // (B,NH,T,HS) scaled by 1/8
    unsigned short* __restrict__ Kb,           // (B,NH,T,HS)
    unsigned short* __restrict__ Vtb) {        // (B,NH,HS,T)
    const int lane = threadIdx.x & 31;
    const int wave = threadIdx.x >> 5;
    const int hl = lane >> 4, l16 = lane & 15;
    const int wm = wave >> 2, wn = wave & 3;
    const int row0 = blockIdx.y * 128;
    const int col0 = blockIdx.x * 128;

    v8f zero = {0.f,0.f,0.f,0.f,0.f,0.f,0.f,0.f};
    v8f acc[4][2] = {{zero, zero}, {zero, zero}, {zero, zero}, {zero, zero}};

    gemm_mainloop(xb, WaT, row0, col0, C_, B_ * T_, N3_, acc);

#pragma unroll
    for (int nj = 0; nj < 2; ++nj) {
        int n = col0 + wn * 32 + 16 * nj + l16;
        float bias = b_attn[n];
        int sec = n >> 10;            // 0=q 1=k 2=v
        int c   = n & (C_ - 1);
        int h   = c >> 6;
        int d   = c & (HS_ - 1);
#pragma unroll
        for (int mi = 0; mi < 4; ++mi) {
#pragma unroll
            for (int rr = 0; rr < 8; ++rr) {
                int bt = row0 + wm * 64 + 16 * mi + rr + 8 * hl;
                int b  = bt >> 11;
                int t  = bt & (T_ - 1);
                float v = acc[mi][nj][rr] + bias;
                size_t bh = (size_t)(b * NH_ + h);
                if (sec == 0)
                    Qb[(bh * T_ + t) * HS_ + d] = f32_to_bf16(v * 0.125f);
                else if (sec == 1)
                    Kb[(bh * T_ + t) * HS_ + d] = f32_to_bf16(v);
                else
                    Vtb[(bh * HS_ + d) * T_ + t] = f32_to_bf16(v);
            }
        }
    }
}

// ---------------- flash attention ----------------
// grid = (T/128, B*NH), block = 256. Wave w owns query tile m0 = (bx*8+w)*16.
__global__ __launch_bounds__(256) void flash_attn(
    const unsigned short* __restrict__ Qb,
    const unsigned short* __restrict__ Kb,
    const unsigned short* __restrict__ Vtb,
    const int* __restrict__ amask,
    unsigned short* __restrict__ Yb) {          // (B,T,C) bf16
    __shared__ unsigned short ldsP[8][16 * 64];

    const int lane = threadIdx.x & 31;
    const int wave = threadIdx.x >> 5;
    const int hl = lane >> 4, l16 = lane & 15;
    const int bh = blockIdx.y;
    const int b  = bh >> 4;
    const int h  = bh & (NH_ - 1);
    const int m0 = (blockIdx.x * 8 + wave) * 16;

    const size_t base = (size_t)bh * T_ * HS_;
    const unsigned short* Q  = Qb  + base;
    const unsigned short* K  = Kb  + base;
    const unsigned short* Vt = Vtb + base;
    const int* mrowp = amask + b * T_;

    FragB16 qa[2];
#pragma unroll
    for (int c = 0; c < 2; ++c)
#pragma unroll
        for (int p = 0; p < 8; ++p) {
            int d = 32 * c + ((p < 4) ? 2 * p : 2 * p + 8) + 8 * hl;
            qa[c].u[p] = *(const unsigned int*)(Q + (size_t)(m0 + l16) * HS_ + d);
        }

    v8f zero = {0.f,0.f,0.f,0.f,0.f,0.f,0.f,0.f};
    v8f o[4] = {zero, zero, zero, zero};
    float mrow[8], lrow[8];
#pragma unroll
    for (int r = 0; r < 8; ++r) { mrow[r] = -1e30f; lrow[r] = 0.f; }

    const int kend = m0 + 16;
    for (int n0 = 0; n0 < kend; n0 += 64) {
        v8f s[4] = {zero, zero, zero, zero};
#pragma unroll
        for (int c = 0; c < 2; ++c) {
#pragma unroll
            for (int j = 0; j < 4; ++j) {
                const unsigned short* Krow = K + (size_t)(n0 + 16 * j + l16) * HS_ + 32 * c;
                FragB16 kb;
#pragma unroll
                for (int p = 0; p < 8; ++p)
                    kb.u[p] = *(const unsigned int*)(Krow + ((p < 4) ? 2 * p : 2 * p + 8) + 8 * hl);
                s[j] = wmma_bf16(qa[c].v, kb.v, s[j]);
            }
        }
        int mv[4];
#pragma unroll
        for (int j = 0; j < 4; ++j) mv[j] = mrowp[n0 + 16 * j + l16];

#pragma unroll
        for (int r = 0; r < 8; ++r) {
            int q = m0 + r + 8 * hl;
            float mx = -1e30f;
#pragma unroll
            for (int j = 0; j < 4; ++j) {
                int key = n0 + 16 * j + l16;
                float sv = s[j][r];
                if (key > q || mv[j] == 0) sv = -1e30f;
                s[j][r] = sv;
                mx = fmaxf(mx, sv);
            }
#pragma unroll
            for (int off = 1; off < 16; off <<= 1)
                mx = fmaxf(mx, __shfl_xor(mx, off, 32));
            float mnew  = fmaxf(mrow[r], mx);
            float alpha = __expf(mrow[r] - mnew);
#pragma unroll
            for (int j = 0; j < 4; ++j) o[j][r] *= alpha;
            float rs = 0.f;
#pragma unroll
            for (int j = 0; j < 4; ++j) {
                float e = __expf(s[j][r] - mnew);
                s[j][r] = e;
                rs += e;
            }
#pragma unroll
            for (int off = 1; off < 16; off <<= 1)
                rs += __shfl_xor(rs, off, 32);
            lrow[r] = lrow[r] * alpha + rs;
            mrow[r] = mnew;
        }

        unsigned short* Pl = ldsP[wave];
#pragma unroll
        for (int j = 0; j < 4; ++j)
#pragma unroll
            for (int r = 0; r < 8; ++r)
                Pl[(r + 8 * hl) * 64 + 16 * j + l16] = f32_to_bf16(s[j][r]);

#pragma unroll
        for (int c = 0; c < 2; ++c) {
            FragB16 pa;
#pragma unroll
            for (int p = 0; p < 8; ++p) {
                int kk = 32 * c + ((p < 4) ? 2 * p : 2 * p + 8) + 8 * hl;
                pa.u[p] = *(const unsigned int*)(Pl + l16 * 64 + kk);
            }
#pragma unroll
            for (int j = 0; j < 4; ++j) {
                const unsigned short* Vcol = Vt + (size_t)(16 * j + l16) * T_ + n0 + 32 * c;
                FragB16 vb;
#pragma unroll
                for (int p = 0; p < 8; ++p)
                    vb.u[p] = *(const unsigned int*)(Vcol + ((p < 4) ? 2 * p : 2 * p + 8) + 8 * hl);
                o[j] = wmma_bf16(pa.v, vb.v, o[j]);
            }
        }
    }

#pragma unroll
    for (int j = 0; j < 4; ++j) {
        int d = 16 * j + l16;
#pragma unroll
        for (int r = 0; r < 8; ++r) {
            int t = m0 + r + 8 * hl;
            float v = o[j][r] / lrow[r];
            Yb[((size_t)b * T_ + t) * C_ + h * HS_ + d] = f32_to_bf16(v);
        }
    }
}

// ---------------- output projection: (4096 x 1024) @ (1024 x 1024) + bias ----------------
__global__ __launch_bounds__(256) void proj_gemm(
    const unsigned short* __restrict__ Yb,
    const unsigned short* __restrict__ WpT,
    const float* __restrict__ b_proj,
    float* __restrict__ out) {
    const int lane = threadIdx.x & 31;
    const int wave = threadIdx.x >> 5;
    const int hl = lane >> 4, l16 = lane & 15;
    const int wm = wave >> 2, wn = wave & 3;
    const int row0 = blockIdx.y * 128;
    const int col0 = blockIdx.x * 128;

    v8f zero = {0.f,0.f,0.f,0.f,0.f,0.f,0.f,0.f};
    v8f acc[4][2] = {{zero, zero}, {zero, zero}, {zero, zero}, {zero, zero}};

    gemm_mainloop(Yb, WpT, row0, col0, C_, B_ * T_, C_, acc);

#pragma unroll
    for (int nj = 0; nj < 2; ++nj) {
        int n = col0 + wn * 32 + 16 * nj + l16;
        float bias = b_proj[n];
#pragma unroll
        for (int mi = 0; mi < 4; ++mi) {
#pragma unroll
            for (int rr = 0; rr < 8; ++rr) {
                int bt = row0 + wm * 64 + 16 * mi + rr + 8 * hl;
                out[(size_t)bt * C_ + n] = acc[mi][nj][rr] + bias;
            }
        }
    }
}

extern "C" void kernel_launch(void* const* d_in, const int* in_sizes, int n_in,
                              void* d_out, int out_size, void* d_ws, size_t ws_size,
                              hipStream_t stream) {
    const float* x      = (const float*)d_in[0];
    const int*   amask  = (const int*)d_in[1];
    const float* W_attn = (const float*)d_in[2];
    const float* b_attn = (const float*)d_in[3];
    const float* W_proj = (const float*)d_in[4];
    const float* b_proj = (const float*)d_in[5];
    float* out = (float*)d_out;

    const size_t NTOK = (size_t)B_ * T_;          // 4096
    const size_t SZX  = NTOK * C_;                // 4,194,304 elems
    const size_t SZWA = (size_t)C_ * N3_;
    const size_t SZWP = (size_t)C_ * C_;

    unsigned short* xb  = (unsigned short*)d_ws;
    unsigned short* WaT = xb  + SZX;
    unsigned short* WpT = WaT + SZWA;
    unsigned short* Qb  = WpT + SZWP;
    unsigned short* Kb  = Qb  + SZX;
    unsigned short* Vtb = Kb  + SZX;
    unsigned short* Yb  = Vtb + SZX;

    conv_f32_bf16<<<dim3((SZX + 255) / 256), 256, 0, stream>>>(x, xb, (int)SZX);
    conv_transpose_bf16<<<dim3((SZWA + 255) / 256), 256, 0, stream>>>(W_attn, WaT, C_, N3_);
    conv_transpose_bf16<<<dim3((SZWP + 255) / 256), 256, 0, stream>>>(W_proj, WpT, C_, C_);

    qkv_gemm<<<dim3(N3_ / 128, NTOK / 128), 256, 0, stream>>>(xb, WaT, b_attn, Qb, Kb, Vtb);

    flash_attn<<<dim3(T_ / 128, B_ * NH_), 256, 0, stream>>>(Qb, Kb, Vtb, amask, Yb);

    proj_gemm<<<dim3(C_ / 128, NTOK / 128), 256, 0, stream>>>(Yb, WpT, b_proj, out);
}
